// ConstrainedLinearChainCRF_55061480735268
// MI455X (gfx1250) — matrix operand order repaired and checked by reading, not compiled, not run on hardware
//
#include <hip/hip_runtime.h>

// ---------------------------------------------------------------------------
// Constrained linear-chain CRF NLL for MI455X (gfx1250, wave32, WMMA).
//
// Forward scan in the scaled-linear domain:
//   w[i,b] = exp(alpha[i,b] - s_b)   (max-normalized, w in [0,1])
//   u[j,b] = sum_i E[i,j] * w[i,b]   with E = exp(trans + tmask)   <-- WMMA
//   wnew   = exp(em[t,j,b]) * u ;  c_b = max_j wnew ; w' = wnew/c ; s += log c
// 16 batches per wave; D = A x B with A = E^T (constant, preloaded in the
// documented 16-bit A layout), B = w (f16, rebuilt each step from LDS).
// ---------------------------------------------------------------------------

typedef __attribute__((ext_vector_type(16))) _Float16 v16h;
typedef __attribute__((ext_vector_type(2)))  __fp16   fp16x2;
typedef __attribute__((ext_vector_type(8)))  float    v8f;
typedef __attribute__((ext_vector_type(4)))  int      i4v;

namespace {
constexpr int Bn = 512;
constexpr int Tn = 2048;
constexpr int Cn = 33;
constexpr int GROUPS = Bn / 16;      // 32 scan workgroups, 16 batches each
constexpr int WROW = 9;              // i4v per batch row in LDS (144B: bank spread)

union V16 { v16h v; _Float16 e[16]; i4v q[2]; };
union V8  { v8f  v; float e[8]; };
union Q8  { i4v  q; _Float16 e[8]; };
}

__device__ __forceinline__ int packh2(float lo, float hi) {
  union { fp16x2 h; int i; } u;
  u.h = __builtin_amdgcn_cvt_pkrtz(lo, hi);   // v_cvt_pk_f16_f32 (RTZ)
  return u.i;
}

#define WMMA_F16(Av, Bv, Cv) \
  __builtin_amdgcn_wmma_f32_16x16x32_f16(false, (Av), false, (Bv), (short)0, (Cv), false, false)

// ---------------------------------------------------------------------------
// Scan kernel: one wave handles 16 consecutive batch rows.
// ---------------------------------------------------------------------------
__global__ __launch_bounds__(32)
void crf_scan_kernel(const float* __restrict__ em, const unsigned char* __restrict__ mask,
                     const float* __restrict__ trans, const float* __restrict__ tmask,
                     const float* __restrict__ starts, const float* __restrict__ smask,
                     const float* __restrict__ ends,   const float* __restrict__ emask,
                     float* __restrict__ outLogZ)
{
  __shared__ i4v ldsW[16 * WROW];          // w[i,b] as f16, b-major, K padded to 64

  const int lane = threadIdx.x;
  const int bl   = lane & 15;              // batch-in-group == lane%16
  const int h    = lane >> 4;              // lane half
  const int batch = blockIdx.x * 16 + bl;

  // ---- constant A tiles: A[m=j, k=i] = E[i][j] = exp(trans+mask), zero padded ----
  // 16-bit A 16x32 layout: lane M = lane%16; VGPR v: K = (v<4 ? 2v : 16+2(v-4)) + 8h, +1
  V16 A[3][2];
  #pragma unroll
  for (int mt = 0; mt < 3; ++mt) {
    const int j = 16 * mt + bl;
    #pragma unroll
    for (int kt = 0; kt < 2; ++kt) {
      #pragma unroll
      for (int v = 0; v < 8; ++v) {
        const int kbase = (v < 4) ? (2 * v) : (16 + 2 * (v - 4));
        const int i0 = 32 * kt + kbase + 8 * h;
        float f0 = 0.f, f1 = 0.f;
        if (j < Cn) {
          if (i0     < Cn) f0 = __expf(trans[i0 * Cn + j]       + tmask[i0 * Cn + j]);
          if (i0 + 1 < Cn) f1 = __expf(trans[(i0 + 1) * Cn + j] + tmask[(i0 + 1) * Cn + j]);
        }
        A[mt][kt].e[2 * v]     = (_Float16)f0;
        A[mt][kt].e[2 * v + 1] = (_Float16)f1;
      }
    }
  }

  const size_t bOff = (size_t)batch * Tn * Cn;
  const float* emA = em + bOff + 8 * h;          // + t*Cn + (16*mt + r)  -> j=16mt+r+8h
  const float* emB = em + bOff;                  // + t*Cn + 32           -> j=32
  const unsigned char* mp = mask + (size_t)batch * Tn;

  double sD;

  // ---- t = 0: alpha0[j] = start[j] + em[b,0,j]; normalize; store f16 to LDS ----
  {
    float a[17];
    float mx = -1e30f;
    #pragma unroll
    for (int k = 0; k < 17; ++k) {
      float av;
      if (k < 8) {                         // j = k + 8h
        const int j = k + 8 * h;
        av = starts[j] + smask[j] + emA[k];
      } else if (k < 16) {                 // j = 16 + (k-8) + 8h
        const int j = 16 + (k - 8) + 8 * h;
        av = starts[j] + smask[j] + emA[16 + (k - 8)];
      } else {                             // j = 32 (valid only on h==0)
        av = (h == 0) ? (starts[32] + smask[32] + emB[32]) : -1e30f;
      }
      a[k] = av;
      mx = fmaxf(mx, av);
    }
    mx = fmaxf(mx, __shfl_xor(mx, 16));
    const int base = bl * WROW + h;
    i4v n0, n1, n2;
    n0.x = packh2(__expf(a[0] - mx), __expf(a[1] - mx));
    n0.y = packh2(__expf(a[2] - mx), __expf(a[3] - mx));
    n0.z = packh2(__expf(a[4] - mx), __expf(a[5] - mx));
    n0.w = packh2(__expf(a[6] - mx), __expf(a[7] - mx));
    n1.x = packh2(__expf(a[8]  - mx), __expf(a[9]  - mx));
    n1.y = packh2(__expf(a[10] - mx), __expf(a[11] - mx));
    n1.z = packh2(__expf(a[12] - mx), __expf(a[13] - mx));
    n1.w = packh2(__expf(a[14] - mx), __expf(a[15] - mx));
    n2.x = packh2(__expf(a[16] - mx), 0.f);
    n2.y = 0; n2.z = 0; n2.w = 0;
    ldsW[base + 0] = n0;                   // j =  0..15 (split by h)
    ldsW[base + 2] = n1;                   // j = 16..31
    ldsW[base + 4] = n2;                   // j = 32..47 (pad = 0)
    if (h == 0) {                          // zero K rows 48..63 once (never rewritten)
      i4v z; z.x = 0; z.y = 0; z.z = 0; z.w = 0;
      ldsW[bl * WROW + 6] = z;
      ldsW[bl * WROW + 7] = z;
    }
    sD = (double)mx;
  }

  // ---- em double-buffer loader (+ prefetch ~8 rows ahead into caches/L2) ----
  auto ldEm = [&](int t, float (&buf)[17], unsigned char& mb) {
    const float* p = emA + (size_t)t * Cn;
    #pragma unroll
    for (int r = 0; r < 8; ++r) buf[r] = p[r];          // j = r + 8h
    #pragma unroll
    for (int r = 0; r < 8; ++r) buf[8 + r] = p[16 + r]; // j = 16 + r + 8h
    buf[16] = emB[(size_t)t * Cn + 32];                 // j = 32
    mb = mp[t];
    const int tp = (t + 8 < Tn) ? (t + 8) : (Tn - 1);
    __builtin_prefetch(emA + (size_t)tp * Cn, 0, 3);        // global_prefetch_b8
    __builtin_prefetch(emA + (size_t)tp * Cn + 16, 0, 3);
  };

  // ---- one recurrence step ----
  auto step = [&](const float (&eb)[17], unsigned char mb) {
    // B tiles (f16 32x16): VGPR v holds K = (v<4 ? 2v : 16+2(v-4)) + 8h, +1
    V16 B0, B1;
    const int lb = bl * WROW + h;
    B0.q[0] = ldsW[lb];     B0.q[1] = ldsW[lb + 2];     // i =  0..31
    B1.q[0] = ldsW[lb + 4]; B1.q[1] = ldsW[lb + 6];     // i = 32..63 (>=33 are 0)

    V8 c0, c1, c2;
    #pragma unroll
    for (int r = 0; r < 8; ++r) { c0.e[r] = 0.f; c1.e[r] = 0.f; c2.e[r] = 0.f; }
    // three independent accumulation chains, interleaved
    c0.v = WMMA_F16(A[0][0].v, B0.v, c0.v);
    c1.v = WMMA_F16(A[1][0].v, B0.v, c1.v);
    c2.v = WMMA_F16(A[2][0].v, B0.v, c2.v);
    c0.v = WMMA_F16(A[0][1].v, B1.v, c0.v);
    c1.v = WMMA_F16(A[1][1].v, B1.v, c1.v);
    c2.v = WMMA_F16(A[2][1].v, B1.v, c2.v);

    // wnew = exp(em) * u ; c = max ; w' = wnew / c
    float w0[8], w1[8], w2;
    #pragma unroll
    for (int r = 0; r < 8; ++r) w0[r] = c0.e[r] * __expf(eb[r]);
    #pragma unroll
    for (int r = 0; r < 8; ++r) w1[r] = c1.e[r] * __expf(eb[8 + r]);
    w2 = c2.e[0] * __expf(eb[16]);      // j=32 on h==0; u==0 on h==1 (pad) -> 0

    float m = w2;
    #pragma unroll
    for (int r = 0; r < 8; ++r) m = fmaxf(m, fmaxf(w0[r], w1[r]));
    m = fmaxf(m, __shfl_xor(m, 16));
    const float inv = 1.0f / m;

    i4v n0, n1, n2;
    n0.x = packh2(w0[0] * inv, w0[1] * inv);
    n0.y = packh2(w0[2] * inv, w0[3] * inv);
    n0.z = packh2(w0[4] * inv, w0[5] * inv);
    n0.w = packh2(w0[6] * inv, w0[7] * inv);
    n1.x = packh2(w1[0] * inv, w1[1] * inv);
    n1.y = packh2(w1[2] * inv, w1[3] * inv);
    n1.z = packh2(w1[4] * inv, w1[5] * inv);
    n1.w = packh2(w1[6] * inv, w1[7] * inv);
    n2.x = packh2(w2 * inv, 0.f);
    n2.y = 0; n2.z = 0; n2.w = 0;

    const int s0 = bl * WROW + h;
    const bool keep = (mb != 0);
    if (__all(keep)) {                       // uniform fast path (mask all ones)
      ldsW[s0 + 0] = n0; ldsW[s0 + 2] = n1; ldsW[s0 + 4] = n2;
      sD += (double)__logf(m);
    } else {                                 // masked step: keep old alpha per batch
      i4v o0 = ldsW[s0 + 0], o1 = ldsW[s0 + 2], o2 = ldsW[s0 + 4];
      i4v r0, r1, r2;
      r0.x = keep ? n0.x : o0.x; r0.y = keep ? n0.y : o0.y;
      r0.z = keep ? n0.z : o0.z; r0.w = keep ? n0.w : o0.w;
      r1.x = keep ? n1.x : o1.x; r1.y = keep ? n1.y : o1.y;
      r1.z = keep ? n1.z : o1.z; r1.w = keep ? n1.w : o1.w;
      r2.x = keep ? n2.x : o2.x; r2.y = keep ? n2.y : o2.y;
      r2.z = keep ? n2.z : o2.z; r2.w = keep ? n2.w : o2.w;
      ldsW[s0 + 0] = r0; ldsW[s0 + 2] = r1; ldsW[s0 + 4] = r2;
      if (keep) sD += (double)__logf(m);
    }
  };

  // ---- main scan: software-pipelined, 2x unrolled so buffers stay in VGPRs ----
  float bufA[17], bufB[17];
  unsigned char mA, mB;
  ldEm(1, bufA, mA);
  for (int t = 1; t < Tn; ) {
    ldEm((t + 1 < Tn) ? t + 1 : Tn - 1, bufB, mB);
    step(bufA, mA);
    ++t;
    if (t >= Tn) break;
    ldEm((t + 1 < Tn) ? t + 1 : Tn - 1, bufA, mA);
    step(bufB, mB);
    ++t;
  }

  // ---- log Z[b] = s_b + log( sum_j w[j,b] * exp(end[j]) ) ----
  {
    const int s0 = bl * WROW + h;
    Q8 r0, r1, r2;
    r0.q = ldsW[s0 + 0]; r1.q = ldsW[s0 + 2]; r2.q = ldsW[s0 + 4];
    float z = 0.f;
    #pragma unroll
    for (int r = 0; r < 8; ++r) {
      const int j = r + 8 * h;
      z += (float)r0.e[r] * __expf(ends[j] + emask[j]);
    }
    #pragma unroll
    for (int r = 0; r < 8; ++r) {
      const int j = 16 + r + 8 * h;
      z += (float)r1.e[r] * __expf(ends[j] + emask[j]);
    }
    if (h == 0) z += (float)r2.e[0] * __expf(ends[32] + emask[32]);
    z += __shfl_xor(z, 16);
    if (lane < 16) outLogZ[batch] = (float)(sD + (double)__logf(z));
  }
}

// ---------------------------------------------------------------------------
// Gold-path score per batch (also warms emissions/tags into the 192MB L2).
// ---------------------------------------------------------------------------
__global__ __launch_bounds__(256)
void crf_score_kernel(const float* __restrict__ em, const int* __restrict__ tags,
                      const unsigned char* __restrict__ mask,
                      const float* __restrict__ trans, const float* __restrict__ tmask,
                      const float* __restrict__ starts, const float* __restrict__ smask,
                      const float* __restrict__ ends,   const float* __restrict__ emask,
                      float* __restrict__ outScore)
{
  __shared__ float sred[256];
  __shared__ int   cred[256];
  const int b = blockIdx.x, tid = threadIdx.x;
  const int* tg = tags + (size_t)b * Tn;
  const unsigned char* mk = mask + (size_t)b * Tn;
  const float* eb = em + (size_t)b * Tn * Cn;

  float part = 0.f;
  int   cnt  = 0;
  for (int t = tid; t < Tn; t += 256) {
    const int mv = mk[t] ? 1 : 0;
    cnt += mv;
    if (t >= 1 && mv) {
      const int tp = tg[t - 1], tc = tg[t];
      part += trans[tp * Cn + tc] + tmask[tp * Cn + tc] + eb[(size_t)t * Cn + tc];
    }
  }
  sred[tid] = part; cred[tid] = cnt;
  __syncthreads();
  for (int s = 128; s > 0; s >>= 1) {
    if (tid < s) { sred[tid] += sred[tid + s]; cred[tid] += cred[tid + s]; }
    __syncthreads();
  }
  if (tid == 0) {
    const int t0 = tg[0];
    float sc = sred[0] + starts[t0] + smask[t0] + eb[t0];
    const int len = cred[0];
    const int lt = tg[len - 1];
    sc += ends[lt] + emask[lt];
    outScore[b] = sc;
  }
}

// ---------------------------------------------------------------------------
// out[0] = mean(logZ - score)
// ---------------------------------------------------------------------------
__global__ __launch_bounds__(512)
void crf_finalize_kernel(const float* __restrict__ logz, const float* __restrict__ score,
                         float* __restrict__ out)
{
  __shared__ float red[512];
  const int tid = threadIdx.x;
  red[tid] = logz[tid] - score[tid];
  __syncthreads();
  for (int s = 256; s > 0; s >>= 1) {
    if (tid < s) red[tid] += red[tid + s];
    __syncthreads();
  }
  if (tid == 0) out[0] = red[0] / (float)Bn;
}

extern "C" void kernel_launch(void* const* d_in, const int* in_sizes, int n_in,
                              void* d_out, int out_size, void* d_ws, size_t ws_size,
                              hipStream_t stream) {
  (void)in_sizes; (void)n_in; (void)out_size; (void)ws_size;
  const float*         em     = (const float*)d_in[0];
  const int*           tags   = (const int*)d_in[1];
  const unsigned char* mask   = (const unsigned char*)d_in[2];
  const float*         trans  = (const float*)d_in[3];
  const float*         starts = (const float*)d_in[4];
  const float*         ends   = (const float*)d_in[5];
  const float*         tmask  = (const float*)d_in[6];
  const float*         smask  = (const float*)d_in[7];
  const float*         emask  = (const float*)d_in[8];

  float* wsLogZ  = (float*)d_ws;
  float* wsScore = wsLogZ + Bn;

  crf_score_kernel<<<Bn, 256, 0, stream>>>(em, tags, mask, trans, tmask,
                                           starts, smask, ends, emask, wsScore);
  crf_scan_kernel<<<GROUPS, 32, 0, stream>>>(em, mask, trans, tmask,
                                             starts, smask, ends, emask, wsLogZ);
  crf_finalize_kernel<<<1, 512, 0, stream>>>(wsLogZ, wsScore, (float*)d_out);
}
